// WindAdvectionBlock_1099511628419
// MI455X (gfx1250) — matrix operand order repaired and checked by reading, not compiled, not run on hardware
//
#include <hip/hip_runtime.h>
#include <cmath>

#define HW    16384
#define WD    128
#define CCH   256
#define MIDC  64
#define NBAT  8

typedef __bf16 bh;
typedef __attribute__((ext_vector_type(16))) __bf16 v16bf;
typedef __attribute__((ext_vector_type(4)))  __bf16 v4bf;
typedef __attribute__((ext_vector_type(8)))  float  v8f;

__device__ __forceinline__ float gelu_exact(float x) {
  return 0.5f * x * (1.0f + erff(x * 0.70710678118654752f));
}

// A fragment (16x32 bf16), tile row-major [m][k], ldk elements per row.
// Lanes 0-15: row m=lane, K = {0..7, 16..23}; lanes 16-31: same rows, K = {8..15, 24..31}.
__device__ __forceinline__ v16bf load_a_frag(const bh* __restrict__ tile, int ldk, int lane) {
  const int m  = lane & 15;
  const int hi = (lane >> 4) & 1;
  const bh* row = tile + m * ldk + hi * 8;
  v16bf a;
#pragma unroll
  for (int h = 0; h < 16; ++h) {
    int k = h + ((h < 8) ? 0 : 8);
    a[h] = row[k];
  }
  return a;
}

// B fragment (32x16 bf16) from a TRANSPOSED tile [n][k], ldk elements per row.
__device__ __forceinline__ v16bf load_b_frag(const bh* __restrict__ tileT, int ldk, int lane) {
  const int n  = lane & 15;
  const int hi = (lane >> 4) & 1;
  const bh* col = tileT + n * ldk + hi * 8;
  v16bf b;
#pragma unroll
  for (int h = 0; h < 16; ++h) b[h] = col[h + ((h < 8) ? 0 : 8)];
  return b;
}

__device__ __forceinline__ v8f wmma_bf16(v16bf a, v16bf b, v8f c) {
  return __builtin_amdgcn_wmma_f32_16x16x32_bf16(false, a, false, b, (short)0, c, false, false);
}

// ---------------------------------------------------------------------------
// Kernel 1: compress 1x1 conv == GEMM  M=B*HW, N=64, K=256
// block: 256 thr (8 waves), 64 pixels x 64 outputs; 8 K-steps of 32
// ---------------------------------------------------------------------------
__global__ __launch_bounds__(256) void compress_kernel(
    const float* __restrict__ feat, const float* __restrict__ w,
    const float* __restrict__ bias, float* __restrict__ xmid) {
  __shared__ alignas(16) bh As[64][32];   // [m][k]
  __shared__ alignas(16) bh Bs[64][32];   // [n][k] (transposed B)
  const int t = threadIdx.x, lane = t & 31, wv = t >> 5;
  const int pix0 = blockIdx.x * 64;
  const int b = pix0 >> 14;           // HW = 2^14
  const int hw0 = pix0 & (HW - 1);
  const int mt0 = (wv >> 2) * 32;     // waves 0-3: rows 0..31, waves 4-7: 32..63
  const int n0  = (wv & 3) * 16;

  v8f acc0 = {}, acc1 = {};
  for (int k0 = 0; k0 < CCH; k0 += 32) {
    __syncthreads();
    // stage A (float4 over contiguous pixels): feat[b, k0+k, hw0+m4..m4+3]
#pragma unroll
    for (int i = t; i < 512; i += 256) {
      int k  = i >> 4;          // 0..31
      int m4 = (i & 15) << 2;   // 0..60
      const float* p = feat + ((size_t)b * CCH + k0 + k) * HW + hw0 + m4;
      __builtin_prefetch(p + 32 * HW, 0, 1);
      float4 f = *(const float4*)p;
      As[m4 + 0][k] = (bh)f.x; As[m4 + 1][k] = (bh)f.y;
      As[m4 + 2][k] = (bh)f.z; As[m4 + 3][k] = (bh)f.w;
    }
    // stage B (float4 contiguous in k): w[n, k0+k4..k4+3]
#pragma unroll
    for (int i = t; i < 512; i += 256) {
      int n  = i >> 3;          // 0..63
      int k4 = (i & 7) << 2;    // 0..28
      float4 f = *(const float4*)(w + (size_t)n * CCH + k0 + k4);
      v4bf pk; pk[0] = (bh)f.x; pk[1] = (bh)f.y; pk[2] = (bh)f.z; pk[3] = (bh)f.w;
      *(v4bf*)&Bs[n][k4] = pk;
    }
    __syncthreads();
    v16bf bf = load_b_frag(&Bs[n0][0], 32, lane);
    v16bf a0 = load_a_frag(&As[mt0][0], 32, lane);
    acc0 = wmma_bf16(a0, bf, acc0);
    v16bf a1 = load_a_frag(&As[mt0 + 16][0], 32, lane);
    acc1 = wmma_bf16(a1, bf, acc1);
  }
  const int n  = n0 + (lane & 15);
  const int mb = mt0 + (((lane >> 4) & 1) ? 8 : 0);
  const float bs = bias[n];
#pragma unroll
  for (int r = 0; r < 8; ++r) {
    xmid[((size_t)b * MIDC + n) * HW + hw0 + mb + r] = acc0[r] + bs;
    xmid[((size_t)b * MIDC + n) * HW + hw0 + mb + 16 + r] = acc1[r] + bs;
  }
}

// ---------------------------------------------------------------------------
// Kernel 2: off1 = 3x3 conv 33->32 SAME + GELU
// ---------------------------------------------------------------------------
__global__ __launch_bounds__(256) void off1_kernel(
    const float* __restrict__ wind, const float* __restrict__ w,
    const float* __restrict__ b, float* __restrict__ h) {
  const int idx = blockIdx.x * 256 + threadIdx.x;   // [B][32][HW]
  const int hw = idx & (HW - 1);
  const int oc = (idx >> 14) & 31;
  const int bb = idx >> 19;
  const int y = hw >> 7, x = hw & 127;
  float s = b[oc];
  for (int ic = 0; ic < 33; ++ic) {
    const float* img = wind + ((size_t)bb * 33 + ic) * HW;
    const float* wk  = w + ((size_t)oc * 33 + ic) * 9;
#pragma unroll
    for (int ky = 0; ky < 3; ++ky) {
      int yy = y + ky - 1;
      if ((unsigned)yy >= (unsigned)WD) continue;
#pragma unroll
      for (int kx = 0; kx < 3; ++kx) {
        int xx = x + kx - 1;
        if ((unsigned)xx >= (unsigned)WD) continue;
        s += wk[ky * 3 + kx] * img[yy * WD + xx];
      }
    }
  }
  h[idx] = gelu_exact(s);
}

// ---------------------------------------------------------------------------
// Kernel 3: off2 = 3x3 conv 32->18 SAME
// ---------------------------------------------------------------------------
__global__ __launch_bounds__(256) void off2_kernel(
    const float* __restrict__ hin, const float* __restrict__ w,
    const float* __restrict__ b, float* __restrict__ offs) {
  const int idx = blockIdx.x * 256 + threadIdx.x;   // [B][18][HW]
  const int hw = idx & (HW - 1);
  const int rest = idx >> 14;
  const int oc = rest % 18;
  const int bb = rest / 18;
  const int y = hw >> 7, x = hw & 127;
  float s = b[oc];
  for (int ic = 0; ic < 32; ++ic) {
    const float* img = hin + ((size_t)bb * 32 + ic) * HW;
    const float* wk  = w + ((size_t)oc * 32 + ic) * 9;
#pragma unroll
    for (int ky = 0; ky < 3; ++ky) {
      int yy = y + ky - 1;
      if ((unsigned)yy >= (unsigned)WD) continue;
#pragma unroll
      for (int kx = 0; kx < 3; ++kx) {
        int xx = x + kx - 1;
        if ((unsigned)xx >= (unsigned)WD) continue;
        s += wk[ky * 3 + kx] * img[yy * WD + xx];
      }
    }
  }
  offs[idx] = s;
}

// ---------------------------------------------------------------------------
// Kernel 3b: pre-pack deform weights, tap-major bf16: wpk[kt][n][c] = dw[n,c,kt]
// Each 8 KB tap slice is exactly the LDS tile image -> async DMA-able.
// ---------------------------------------------------------------------------
__global__ __launch_bounds__(256) void pack_dw_kernel(
    const float* __restrict__ dw, bh* __restrict__ wpk) {
  int i = blockIdx.x * 256 + threadIdx.x;    // 9*64*64 = 36864
  int kt = i >> 12;
  int n  = (i >> 6) & 63;
  int c  = i & 63;
  wpk[i] = (bh)dw[(size_t)n * 576 + c * 9 + kt];
}

__device__ __forceinline__ float samp(const float* __restrict__ img, int y, int x) {
  bool ok = (y >= 0) && (y < WD) && (x >= 0) && (x < WD);
  int yc = min(max(y, 0), WD - 1);
  int xc = min(max(x, 0), WD - 1);
  float v = img[yc * WD + xc];
  return ok ? v : 0.0f;
}

// ---------------------------------------------------------------------------
// Kernel 4: deformable conv == gather + GEMM  M=B*HW, N=64, K=576 (tap-major)
// Per-tap weight tile fetched with async global->LDS DMA (ASYNCcnt),
// overlapped with bilinear coordinate setup.
// ---------------------------------------------------------------------------
__global__ __launch_bounds__(256) void deform_kernel(
    const float* __restrict__ xmid, const float* __restrict__ offs,
    const bh* __restrict__ wpk, const float* __restrict__ db,
    float* __restrict__ x2) {
  __shared__ alignas(16) bh Wt[64][64];   // per-tap B tile transposed: [n][c]  (8 KB)
  __shared__ alignas(16) bh As[64][32];   // sampled A tile [m][c-half]
  __shared__ int   iy[64], ix[64];
  __shared__ float fy[64], fx[64];
  const int t = threadIdx.x, lane = t & 31, wv = t >> 5;
  const int pix0 = blockIdx.x * 64;
  const int b = pix0 >> 14;
  const int hw0 = pix0 & (HW - 1);
  const int mt0 = (wv >> 2) * 32;
  const int n0  = (wv & 3) * 16;
  const unsigned ldsW = (unsigned)(uintptr_t)(&Wt[0][0]);

  v8f acc0 = {}, acc1 = {};
  for (int kt = 0; kt < 9; ++kt) {
    __syncthreads();
    // async DMA the 8 KB per-tap weight tile into LDS (2 x b128 per lane)
    {
      const bh* gsrc = wpk + (size_t)kt * 4096;
      unsigned go = (unsigned)t * 16u;
      asm volatile("global_load_async_to_lds_b128 %0, %1, %2"
                   :: "v"(ldsW + go), "v"(go), "s"(gsrc) : "memory");
      asm volatile("global_load_async_to_lds_b128 %0, %1, %2"
                   :: "v"(ldsW + go + 4096u), "v"(go + 4096u), "s"(gsrc) : "memory");
    }
    // bilinear coordinates for this tap (overlaps the DMA)
    if (t < 64) {
      int hw = hw0 + t;
      int y = hw >> 7, x = hw & 127;
      float dy = offs[((size_t)b * 18 + kt * 2 + 0) * HW + hw];
      float dx = offs[((size_t)b * 18 + kt * 2 + 1) * HW + hw];
      float ys = (float)y + (float)(kt / 3 - 1) + dy;
      float xs = (float)x + (float)(kt % 3 - 1) + dx;
      float y0 = floorf(ys), x0 = floorf(xs);
      iy[t] = (int)y0; ix[t] = (int)x0;
      fy[t] = ys - y0; fx[t] = xs - x0;
    }
    asm volatile("s_wait_asynccnt 0" ::: "memory");
    __syncthreads();
    for (int ch = 0; ch < 2; ++ch) {
      const int cbase = ch * 32;
      // bilinear-sample A tile: 64 pixels x 32 channels
      for (int i = t; i < 64 * 32; i += 256) {
        int m = i >> 5;
        int c = (i & 31) + cbase;
        int y0 = iy[m], x0 = ix[m];
        float wy = fy[m], wx = fx[m];
        const float* img = xmid + ((size_t)b * MIDC + c) * HW;
        float v00 = samp(img, y0, x0);
        float v01 = samp(img, y0, x0 + 1);
        float v10 = samp(img, y0 + 1, x0);
        float v11 = samp(img, y0 + 1, x0 + 1);
        float v = v00 * (1.f - wy) * (1.f - wx) + v01 * (1.f - wy) * wx
                + v10 * wy * (1.f - wx)         + v11 * wy * wx;
        As[m][i & 31] = (bh)v;
      }
      __syncthreads();
      v16bf bf = load_b_frag(&Wt[n0][cbase], 64, lane);
      v16bf a0 = load_a_frag(&As[mt0][0], 32, lane);
      acc0 = wmma_bf16(a0, bf, acc0);
      v16bf a1 = load_a_frag(&As[mt0 + 16][0], 32, lane);
      acc1 = wmma_bf16(a1, bf, acc1);
      __syncthreads();
    }
  }
  const int n  = n0 + (lane & 15);
  const int mb = mt0 + (((lane >> 4) & 1) ? 8 : 0);
  const float bias = db[n];
#pragma unroll
  for (int r = 0; r < 8; ++r) {
    x2[((size_t)b * MIDC + n) * HW + hw0 + mb + r] = acc0[r] + bias;
    x2[((size_t)b * MIDC + n) * HW + hw0 + mb + 16 + r] = acc1[r] + bias;
  }
}

// ---------------------------------------------------------------------------
// Kernel 5: expand 1x1 conv == GEMM  M=B*HW, N=256, K=64; fused GN partial sums
// ---------------------------------------------------------------------------
__global__ __launch_bounds__(256) void expand_kernel(
    const float* __restrict__ x2, const float* __restrict__ ew,
    const float* __restrict__ eb, float* __restrict__ y,
    float* __restrict__ gsum, float* __restrict__ gsumsq) {
  __shared__ alignas(16) bh Bs[256][64];  // [n][k] transposed (32 KB)
  __shared__ alignas(16) bh As[64][64];   // [m][k] (8 KB)
  const int t = threadIdx.x, lane = t & 31, wv = t >> 5;
  const int pix0 = blockIdx.x * 64;
  const int b = pix0 >> 14;
  const int hw0 = pix0 & (HW - 1);

  // stage B (float4 contiguous in k)
#pragma unroll
  for (int i = t; i < 4096; i += 256) {
    int n  = i >> 4;          // 0..255
    int k4 = (i & 15) << 2;   // 0..60
    float4 f = *(const float4*)(ew + (size_t)n * MIDC + k4);
    v4bf pk; pk[0] = (bh)f.x; pk[1] = (bh)f.y; pk[2] = (bh)f.z; pk[3] = (bh)f.w;
    *(v4bf*)&Bs[n][k4] = pk;
  }
  // stage A (float4 over contiguous pixels)
#pragma unroll
  for (int i = t; i < 1024; i += 256) {
    int k  = i & 63;
    int m4 = (i >> 6) << 2;   // 0..60
    const float* p = x2 + ((size_t)b * MIDC + k) * HW + hw0 + m4;
    float4 f = *(const float4*)p;
    As[m4 + 0][k] = (bh)f.x; As[m4 + 1][k] = (bh)f.y;
    As[m4 + 2][k] = (bh)f.z; As[m4 + 3][k] = (bh)f.w;
  }
  __syncthreads();

  const int n0 = wv * 32;            // wave covers n in [32*wv, 32*wv+32)
  v8f acc[4][2];
#pragma unroll
  for (int mt = 0; mt < 4; ++mt)
#pragma unroll
    for (int nt = 0; nt < 2; ++nt) acc[mt][nt] = (v8f){};

#pragma unroll
  for (int ks = 0; ks < 2; ++ks) {
    const int k0 = ks * 32;
    v16bf af[4];
#pragma unroll
    for (int mt = 0; mt < 4; ++mt) af[mt] = load_a_frag(&As[mt * 16][k0], 64, lane);
#pragma unroll
    for (int nt = 0; nt < 2; ++nt) {
      v16bf bf = load_b_frag(&Bs[n0 + nt * 16][k0], 64, lane);
#pragma unroll
      for (int mt = 0; mt < 4; ++mt) acc[mt][nt] = wmma_bf16(af[mt], bf, acc[mt][nt]);
    }
  }

  float psum = 0.f, psumsq = 0.f;
  const int nlo = lane & 15;
  const int mhi = ((lane >> 4) & 1) ? 8 : 0;
#pragma unroll
  for (int nt = 0; nt < 2; ++nt) {
    const int n = n0 + nt * 16 + nlo;
    const float bias = eb[n];
#pragma unroll
    for (int mt = 0; mt < 4; ++mt) {
#pragma unroll
      for (int r = 0; r < 8; ++r) {
        int m = mt * 16 + mhi + r;
        float v = acc[mt][nt][r] + bias;
        y[((size_t)b * CCH + n) * HW + hw0 + m] = v;
        psum += v; psumsq += v * v;
      }
    }
  }
  // wave32 reduction, one atomic per wave per stat (group = n0/64 = wv>>1)
#pragma unroll
  for (int o = 16; o > 0; o >>= 1) {
    psum   += __shfl_xor(psum, o, 32);
    psumsq += __shfl_xor(psumsq, o, 32);
  }
  if (lane == 0) {
    int g = wv >> 1;
    atomicAdd(&gsum[b * 4 + g], psum);
    atomicAdd(&gsumsq[b * 4 + g], psumsq);
  }
}

// ---------------------------------------------------------------------------
// Stats helpers
// ---------------------------------------------------------------------------
__global__ void zero_stats_kernel(float* __restrict__ stats) {
  if (threadIdx.x < 128) stats[threadIdx.x] = 0.f;
}

__global__ void gn_stats_kernel(const float* __restrict__ gsum,
                                const float* __restrict__ gsumsq,
                                float* __restrict__ mean, float* __restrict__ rstd) {
  int i = threadIdx.x;
  if (i < 32) {
    const float N = 64.0f * (float)HW;
    float m = gsum[i] / N;
    float v = gsumsq[i] / N - m * m;
    mean[i] = m;
    rstd[i] = rsqrtf(v + 1e-5f);
  }
}

// ---------------------------------------------------------------------------
// Kernel 6: normalize + GELU + residual, float4, in place on d_out
// ---------------------------------------------------------------------------
__global__ __launch_bounds__(256) void finalize_kernel(
    const float* __restrict__ feat, const float* __restrict__ mean,
    const float* __restrict__ rstd, const float* __restrict__ gw,
    const float* __restrict__ gb, float* __restrict__ out) {
  const int i4 = blockIdx.x * 256 + threadIdx.x;   // index in float4 units
  const int idx = i4 << 2;                          // [B][256][HW]
  const int c = (idx >> 14) & 255;
  const int b = idx >> 22;
  const int g = c >> 6;
  const float mu = mean[b * 4 + g];
  const float rs = rstd[b * 4 + g];
  const float sw = gw[c] * rs;
  const float sb = gb[c];
  float4 yv = ((const float4*)out)[i4];
  float4 fv = ((const float4*)feat)[i4];
  float4 o;
  o.x = fv.x + gelu_exact((yv.x - mu) * sw + sb);
  o.y = fv.y + gelu_exact((yv.y - mu) * sw + sb);
  o.z = fv.z + gelu_exact((yv.z - mu) * sw + sb);
  o.w = fv.w + gelu_exact((yv.w - mu) * sw + sb);
  ((float4*)out)[i4] = o;
}

// ---------------------------------------------------------------------------
extern "C" void kernel_launch(void* const* d_in, const int* in_sizes, int n_in,
                              void* d_out, int out_size, void* d_ws, size_t ws_size,
                              hipStream_t stream) {
  (void)in_sizes; (void)n_in; (void)out_size; (void)ws_size;
  const float* feat = (const float*)d_in[0];
  const float* wind = (const float*)d_in[1];
  const float* cw   = (const float*)d_in[2];
  const float* cb   = (const float*)d_in[3];
  const float* o1w  = (const float*)d_in[4];
  const float* o1b  = (const float*)d_in[5];
  const float* o2w  = (const float*)d_in[6];
  const float* o2b  = (const float*)d_in[7];
  const float* dw   = (const float*)d_in[8];
  const float* db   = (const float*)d_in[9];
  const float* ew   = (const float*)d_in[10];
  const float* eb   = (const float*)d_in[11];
  const float* gw   = (const float*)d_in[12];
  const float* gb   = (const float*)d_in[13];
  float* out = (float*)d_out;

  float* ws    = (float*)d_ws;
  float* xmid  = ws;                                   // 8*64*HW
  float* hbuf  = xmid + (size_t)NBAT * MIDC * HW;      // 8*32*HW
  float* offs  = hbuf + (size_t)NBAT * 32 * HW;        // 8*18*HW
  float* x2    = offs + (size_t)NBAT * 18 * HW;        // 8*64*HW
  float* stats = x2 + (size_t)NBAT * MIDC * HW;        // 128 floats
  bh*    wpk   = (bh*)(stats + 128);                   // 9*64*64 bf16 (16B aligned)

  hipLaunchKernelGGL(zero_stats_kernel, dim3(1), dim3(128), 0, stream, stats);
  hipLaunchKernelGGL(compress_kernel, dim3(2048), dim3(256), 0, stream, feat, cw, cb, xmid);
  hipLaunchKernelGGL(off1_kernel, dim3(16384), dim3(256), 0, stream, wind, o1w, o1b, hbuf);
  hipLaunchKernelGGL(off2_kernel, dim3(9216), dim3(256), 0, stream, hbuf, o2w, o2b, offs);
  hipLaunchKernelGGL(pack_dw_kernel, dim3(144), dim3(256), 0, stream, dw, wpk);
  hipLaunchKernelGGL(deform_kernel, dim3(2048), dim3(256), 0, stream, xmid, offs, wpk, db, x2);
  hipLaunchKernelGGL(expand_kernel, dim3(2048), dim3(256), 0, stream, x2, ew, eb, out,
                     stats, stats + 32);
  hipLaunchKernelGGL(gn_stats_kernel, dim3(1), dim3(32), 0, stream,
                     stats, stats + 32, stats + 64, stats + 96);
  hipLaunchKernelGGL(finalize_kernel, dim3(32768), dim3(256), 0, stream,
                     feat, stats + 64, stats + 96, gw, gb, out);
}
